// VQVAE_73710228734088
// MI455X (gfx1250) — matrix operand order repaired and checked by reading, not compiled, not run on hardware
//
#include <hip/hip_runtime.h>
#include <hip/hip_bf16.h>

// ---------------- CDNA5 WMMA types ----------------
typedef __attribute__((ext_vector_type(16))) __bf16 v16bf;
typedef __attribute__((ext_vector_type(8)))  float  v8f;

union Frag {
    v16bf v;
    uint4 q[2];
    unsigned short u[16];
};

__device__ __forceinline__ unsigned short f2bf(float f) {
    unsigned int u = __float_as_uint(f);
    return (unsigned short)((u + 0x7FFFu + ((u >> 16) & 1u)) >> 16);
}
__device__ __forceinline__ float bf2f(unsigned short h) {
    return __uint_as_float(((unsigned int)h) << 16);
}

// =======================================================================
// Weight / codebook packing kernels (fp32 -> bf16, layout permutes)
// =======================================================================

// OIHW [COUT][CIN][3][3] -> [9][COUT][CIN]  (encoder conv)
__global__ void pack_conv_enc_k(const float* __restrict__ w, unsigned short* __restrict__ out,
                                int CIN, int COUT) {
    int total = 9 * CIN * COUT;
    for (int t = blockIdx.x * blockDim.x + threadIdx.x; t < total; t += gridDim.x * blockDim.x) {
        int i = t % CIN; int rest = t / CIN;
        int o = rest % COUT; int tap = rest / COUT;
        out[t] = f2bf(w[(o * CIN + i) * 9 + tap]);
    }
}

// [CIN][COUT][3][3] (convT weight) -> [9][COUT][CIN] with spatial flip
__global__ void pack_conv_dec_k(const float* __restrict__ w, unsigned short* __restrict__ out,
                                int CIN, int COUT) {
    int total = 9 * CIN * COUT;
    for (int t = blockIdx.x * blockDim.x + threadIdx.x; t < total; t += gridDim.x * blockDim.x) {
        int i = t % CIN; int rest = t / CIN;
        int o = rest % COUT; int tap = rest / COUT;
        out[t] = f2bf(w[(i * COUT + o) * 9 + (8 - tap)]);
    }
}

// enc_fc_w [128][1152 (c*9+s)] -> Bt [N=128][K=1152 (s*128+c)]
__global__ void pack_enc_fc_k(const float* __restrict__ w, unsigned short* __restrict__ out) {
    int total = 128 * 1152;
    for (int t = blockIdx.x * blockDim.x + threadIdx.x; t < total; t += gridDim.x * blockDim.x) {
        int k = t % 1152; int n = t / 1152;
        int s = k / 128;  int c = k % 128;
        out[t] = f2bf(w[n * 1152 + c * 9 + s]);
    }
}

// codebook [1024][128] -> bf16 [1024][128]
__global__ void pack_cb_k(const float* __restrict__ cb, unsigned short* __restrict__ out) {
    int total = 1024 * 128;
    for (int t = blockIdx.x * blockDim.x + threadIdx.x; t < total; t += gridDim.x * blockDim.x)
        out[t] = f2bf(cb[t]);
}

// 0.5 * ||codebook_n||^2
__global__ void cnorm_k(const float* __restrict__ cb, float* __restrict__ cnh) {
    int n = blockIdx.x * blockDim.x + threadIdx.x;
    if (n >= 1024) return;
    float s = 0.f;
    for (int k = 0; k < 128; ++k) { float v = cb[n * 128 + k]; s += v * v; }
    cnh[n] = 0.5f * s;
}

// dec_fc_w [1152 (c*9+s)][128] -> Bt [N=1152 (s*128+c)][K=128]
__global__ void pack_dec_fc_k(const float* __restrict__ w, unsigned short* __restrict__ out) {
    int total = 1152 * 128;
    for (int t = blockIdx.x * blockDim.x + threadIdx.x; t < total; t += gridDim.x * blockDim.x) {
        int k = t % 128; int n = t / 128;
        int s = n / 128; int c = n % 128;
        out[t] = f2bf(w[(c * 9 + s) * 128 + k]);
    }
}

// dec_fc_b permute into n' = s*128+c ordering
__global__ void pack_dec_bias_k(const float* __restrict__ b, float* __restrict__ bp) {
    int n = blockIdx.x * blockDim.x + threadIdx.x;
    if (n >= 1152) return;
    bp[n] = b[(n % 128) * 9 + n / 128];
}

// =======================================================================
// conv1: 1->32 ch on 24x24, relu, maxpool2 -> NHWC bf16 [B][12][12][32]
// =======================================================================
__global__ __launch_bounds__(256)
void conv1_relu_pool(const float* __restrict__ x, const float* __restrict__ w,
                     const float* __restrict__ b, unsigned short* __restrict__ out, int B) {
    __shared__ float sw[288];
    __shared__ float sb[32];
    for (int t = threadIdx.x; t < 288; t += 256) sw[t] = w[t];
    for (int t = threadIdx.x; t < 32; t += 256) sb[t] = b[t];
    __syncthreads();
    int gid = blockIdx.x * 256 + threadIdx.x;
    if (gid >= B * 144) return;
    int bimg = gid / 144; int p = gid % 144;
    int oy = p / 12, ox = p % 12;
    float win[4][4];
    const float* xb = x + (size_t)bimg * 576;
#pragma unroll
    for (int i = 0; i < 4; ++i) {
        int yy = 2 * oy - 1 + i;
#pragma unroll
        for (int j = 0; j < 4; ++j) {
            int xx = 2 * ox - 1 + j;
            win[i][j] = (yy >= 0 && yy < 24 && xx >= 0 && xx < 24) ? xb[yy * 24 + xx] : 0.f;
        }
    }
    unsigned short* op = out + (size_t)gid * 32;
    for (int o = 0; o < 32; ++o) {
        float m = -3.0e38f;
#pragma unroll
        for (int i = 0; i < 2; ++i)
#pragma unroll
            for (int j = 0; j < 2; ++j) {
                float s = sb[o];
#pragma unroll
                for (int dy = 0; dy < 3; ++dy)
#pragma unroll
                    for (int dx = 0; dx < 3; ++dx)
                        s += win[i + dy][j + dx] * sw[o * 9 + dy * 3 + dx];
                m = fmaxf(m, s);
            }
        op[o] = f2bf(fmaxf(m, 0.f));
    }
}

// =======================================================================
// Implicit-GEMM 3x3 conv (pad 1, stride 1) + bias + relu + pool2/upsample2
// NHWC bf16 in/out.  Weights packed [9][COUT][CIN] bf16.
// Block = 128 threads (4 waves). Each block handles IMGS images.
// =======================================================================
template<int CIN, int COUT, int S, int IMGS, bool POOL>
__global__ __launch_bounds__(128)
void conv3x3_relu_wmma(const unsigned short* __restrict__ in,
                       const unsigned short* __restrict__ wp,
                       const float* __restrict__ bias,
                       unsigned short* __restrict__ out) {
    constexpr int PAD = S + 2;
    constexpr int M   = IMGS * S * S;
    constexpr int MT  = (M + 15) / 16;
    constexpr int NT  = COUT / 16;
    constexpr int NPH = (NT + 3) / 4;
    constexpr int OST = (COUT < 64) ? COUT : 64;

    __shared__ __align__(16) unsigned short sIn[IMGS * PAD * PAD * CIN];
    __shared__ __align__(16) unsigned short sOut[MT * 16 * OST];

    const int tid = threadIdx.x;
    const int b0  = blockIdx.x * IMGS;
    const int lane = tid & 31, wave = tid >> 5;
    const int h = lane >> 4, l15 = lane & 15;

    // ---- cooperative haloed NHWC tile load (zero padding) ----
    constexpr int INW = IMGS * PAD * PAD * (CIN / 8);
    for (int t = tid; t < INW; t += 128) {
        int c8 = t % (CIN / 8); int rest = t / (CIN / 8);
        int px = rest % PAD; rest /= PAD;
        int py = rest % PAD; int img = rest / PAD;
        uint4 val = {0u, 0u, 0u, 0u};
        if (py >= 1 && py <= S && px >= 1 && px <= S) {
            const unsigned short* p =
                in + ((size_t)(b0 + img) * S * S + (size_t)(py - 1) * S + (px - 1)) * CIN + c8 * 8;
            val = *(const uint4*)p;
        }
        *(uint4*)(&sIn[(size_t)t * 8]) = val;
    }
    __syncthreads();

    for (int ph = 0; ph < NPH; ++ph) {
        const int nt = ph * 4 + wave;
        if (nt < NT) {
            const float bn = bias[nt * 16 + l15];
            for (int mt = 0; mt < MT; ++mt) {
                v8f acc = {};
                int m = mt * 16 + l15; if (m >= M) m = M - 1;
                const int img = m / (S * S); const int p = m % (S * S);
                const int y = p / S, x = p % S;
                const unsigned short* arow = &sIn[((size_t)(img * PAD + y) * PAD + x) * CIN];
#pragma unroll
                for (int tap = 0; tap < 9; ++tap) {
                    const int dy = tap / 3, dx = tap % 3;
                    const unsigned short* apix = arow + ((size_t)dy * PAD + dx) * CIN;
                    const unsigned short* brow =
                        wp + (size_t)tap * COUT * CIN + (size_t)(nt * 16 + l15) * CIN;
#pragma unroll
                    for (int ks = 0; ks < CIN / 32; ++ks) {
                        Frag Af, Bf;
                        Af.q[0] = *(const uint4*)(apix + ks * 32 + h * 8);
                        Af.q[1] = *(const uint4*)(apix + ks * 32 + h * 8 + 16);
                        Bf.q[0] = *(const uint4*)(brow + ks * 32 + h * 8);
                        Bf.q[1] = *(const uint4*)(brow + ks * 32 + h * 8 + 16);
                        acc = __builtin_amdgcn_wmma_f32_16x16x32_bf16(
                            false, Af.v, false, Bf.v, (short)0, acc, false, false);
                    }
                }
                const int cb = (nt - ph * 4) * 16 + l15;
#pragma unroll
                for (int r = 0; r < 8; ++r) {
                    int mm = mt * 16 + r + 8 * h;
                    float v = acc[r] + bn;
                    v = v > 0.f ? v : 0.f;
                    if (mm < M) sOut[(size_t)mm * OST + cb] = f2bf(v);
                }
            }
        }
        __syncthreads();

        const int cc = ((COUT - ph * 64) < 64) ? (COUT - ph * 64) : 64;
        if (POOL) {
            constexpr int OS = S / 2;
            const int tot = IMGS * OS * OS * cc;
            for (int t = tid; t < tot; t += 128) {
                int c = t % cc; int rest = t / cc;
                int ox = rest % OS; rest /= OS;
                int oy = rest % OS; int img = rest / OS;
                const unsigned short* q =
                    &sOut[((size_t)(img * S * S) + (size_t)(2 * oy) * S + 2 * ox) * OST + c];
                float v0 = bf2f(q[0]);
                float v1 = bf2f(q[OST]);
                float v2 = bf2f(q[(size_t)S * OST]);
                float v3 = bf2f(q[(size_t)S * OST + OST]);
                float v = fmaxf(fmaxf(v0, v1), fmaxf(v2, v3));
                out[((size_t)(b0 + img) * OS * OS + (size_t)oy * OS + ox) * COUT + ph * 64 + c] =
                    f2bf(v);
            }
        } else {
            constexpr int OS2 = 2 * S;
            const int tot = IMGS * S * S * cc;
            for (int t = tid; t < tot; t += 128) {
                int c = t % cc; int rest = t / cc;
                int x = rest % S; rest /= S;
                int y = rest % S; int img = rest / S;
                unsigned short v = sOut[((size_t)(img * S * S) + (size_t)y * S + x) * OST + c];
                size_t base = (size_t)(b0 + img) * OS2 * OS2;
                int oc = ph * 64 + c;
                out[(base + (size_t)(2 * y) * OS2 + 2 * x) * COUT + oc] = v;
                out[(base + (size_t)(2 * y) * OS2 + 2 * x + 1) * COUT + oc] = v;
                out[(base + (size_t)(2 * y + 1) * OS2 + 2 * x) * COUT + oc] = v;
                out[(base + (size_t)(2 * y + 1) * OS2 + 2 * x + 1) * COUT + oc] = v;
            }
        }
        __syncthreads();
    }
}

// =======================================================================
// Generic FC GEMM: out[M][N] = A[M][K] * Bt[N][K]^T + bias   (bf16 in/out)
// Block = 4 waves, each wave owns a 16-row M-tile.
// =======================================================================
__global__ __launch_bounds__(128)
void fc_wmma(const unsigned short* __restrict__ A, const unsigned short* __restrict__ Bt,
             const float* __restrict__ bias, unsigned short* __restrict__ out,
             int K, int N) {
    const int tid = threadIdx.x, lane = tid & 31, wave = tid >> 5;
    const int h = lane >> 4, l15 = lane & 15;
    const int mbase = (blockIdx.x * 4 + wave) * 16;
    const unsigned short* arow = A + (size_t)(mbase + l15) * K;
    const int NT = N >> 4, KS = K >> 5;
    for (int nt = 0; nt < NT; ++nt) {
        v8f acc = {};
        const unsigned short* brow = Bt + (size_t)(nt * 16 + l15) * K;
        for (int ks = 0; ks < KS; ++ks) {
            Frag Af, Bf;
            Af.q[0] = *(const uint4*)(arow + ks * 32 + h * 8);
            Af.q[1] = *(const uint4*)(arow + ks * 32 + h * 8 + 16);
            Bf.q[0] = *(const uint4*)(brow + ks * 32 + h * 8);
            Bf.q[1] = *(const uint4*)(brow + ks * 32 + h * 8 + 16);
            acc = __builtin_amdgcn_wmma_f32_16x16x32_bf16(
                false, Af.v, false, Bf.v, (short)0, acc, false, false);
        }
        const int n = nt * 16 + l15;
        const float bn = bias ? bias[n] : 0.f;
#pragma unroll
        for (int r = 0; r < 8; ++r)
            out[(size_t)(mbase + r + 8 * h) * N + n] = f2bf(acc[r] + bn);
    }
}

// =======================================================================
// VQ: fused z@cb^T GEMM + argmin over 1024 codewords
// score = 0.5*||c_n||^2 - z . c_n  (z-norm dropped: constant per row)
// =======================================================================
__global__ __launch_bounds__(128)
void vq_argmin_wmma(const unsigned short* __restrict__ Z, const unsigned short* __restrict__ CB,
                    const float* __restrict__ cnh, int* __restrict__ idx) {
    const int tid = threadIdx.x, lane = tid & 31, wave = tid >> 5;
    const int h = lane >> 4, l15 = lane & 15;
    const int mbase = (blockIdx.x * 4 + wave) * 16;
    float bestv[8]; int besti[8];
#pragma unroll
    for (int r = 0; r < 8; ++r) { bestv[r] = 3.0e38f; besti[r] = 0; }
    const unsigned short* arow = Z + (size_t)(mbase + l15) * 128;
    for (int nt = 0; nt < 64; ++nt) {
        v8f acc = {};
        const unsigned short* brow = CB + (size_t)(nt * 16 + l15) * 128;
#pragma unroll
        for (int ks = 0; ks < 4; ++ks) {
            Frag Af, Bf;
            Af.q[0] = *(const uint4*)(arow + ks * 32 + h * 8);
            Af.q[1] = *(const uint4*)(arow + ks * 32 + h * 8 + 16);
            Bf.q[0] = *(const uint4*)(brow + ks * 32 + h * 8);
            Bf.q[1] = *(const uint4*)(brow + ks * 32 + h * 8 + 16);
            acc = __builtin_amdgcn_wmma_f32_16x16x32_bf16(
                false, Af.v, false, Bf.v, (short)0, acc, false, false);
        }
        const int n = nt * 16 + l15;
        const float c = cnh[n];
#pragma unroll
        for (int r = 0; r < 8; ++r) {
            float s = c - acc[r];
            if (s < bestv[r]) { bestv[r] = s; besti[r] = n; }
        }
    }
#pragma unroll
    for (int r = 0; r < 8; ++r) {
#pragma unroll
        for (int msk = 1; msk < 16; msk <<= 1) {
            float ov = __shfl_xor(bestv[r], msk, 32);
            int   oi = __shfl_xor(besti[r], msk, 32);
            if (ov < bestv[r] || (ov == bestv[r] && oi < besti[r])) { bestv[r] = ov; besti[r] = oi; }
        }
        if (l15 == 0) idx[mbase + r + 8 * h] = besti[r];
    }
}

__global__ void vq_gather(const int* __restrict__ idx, const float* __restrict__ cb,
                          unsigned short* __restrict__ zq, int total) {
    int t = blockIdx.x * 256 + threadIdx.x;
    if (t >= total) return;
    int b = t >> 7, k = t & 127;
    zq[t] = f2bf(cb[(size_t)idx[b] * 128 + k]);
}

// =======================================================================
// dct3: 32->1 convT on 12x12, relu, upsample2 -> fp32 [B][24][24]
// =======================================================================
__global__ __launch_bounds__(256)
void dct3_relu_upsample(const unsigned short* __restrict__ in, const float* __restrict__ w,
                        const float* __restrict__ b, float* __restrict__ out, int B) {
    __shared__ float sw[288]; // [tap][c], flipped
    for (int t = threadIdx.x; t < 288; t += 256) {
        int tap = t / 32, c = t % 32;
        sw[t] = w[c * 9 + (8 - tap)];
    }
    __syncthreads();
    float bias = b[0];
    int gid = blockIdx.x * 256 + threadIdx.x;
    if (gid >= B * 144) return;
    int bimg = gid / 144; int p = gid % 144;
    int y = p / 12, x = p % 12;
    const unsigned short* ib = in + (size_t)bimg * 144 * 32;
    float s = bias;
#pragma unroll
    for (int tap = 0; tap < 9; ++tap) {
        int yy = y + tap / 3 - 1, xx = x + tap % 3 - 1;
        if (yy >= 0 && yy < 12 && xx >= 0 && xx < 12) {
            const unsigned short* q = ib + ((size_t)yy * 12 + xx) * 32;
#pragma unroll
            for (int c = 0; c < 32; ++c) s += bf2f(q[c]) * sw[tap * 32 + c];
        }
    }
    s = fmaxf(s, 0.f);
    float* ob = out + (size_t)bimg * 576;
    ob[(2 * y) * 24 + 2 * x] = s;     ob[(2 * y) * 24 + 2 * x + 1] = s;
    ob[(2 * y + 1) * 24 + 2 * x] = s; ob[(2 * y + 1) * 24 + 2 * x + 1] = s;
}

// =======================================================================
// Host launch
// =======================================================================
extern "C" void kernel_launch(void* const* d_in, const int* in_sizes, int n_in,
                              void* d_out, int out_size, void* d_ws, size_t ws_size,
                              hipStream_t stream) {
    const float* x        = (const float*)d_in[0];
    const float* conv1_w  = (const float*)d_in[1];
    const float* conv1_b  = (const float*)d_in[2];
    const float* conv2_w  = (const float*)d_in[3];
    const float* conv2_b  = (const float*)d_in[4];
    const float* conv3_w  = (const float*)d_in[5];
    const float* conv3_b  = (const float*)d_in[6];
    const float* enc_fc_w = (const float*)d_in[7];
    const float* enc_fc_b = (const float*)d_in[8];
    const float* codebook = (const float*)d_in[9];
    const float* dec_fc_w = (const float*)d_in[10];
    const float* dec_fc_b = (const float*)d_in[11];
    const float* dct1_w   = (const float*)d_in[12];
    const float* dct1_b   = (const float*)d_in[13];
    const float* dct2_w   = (const float*)d_in[14];
    const float* dct2_b   = (const float*)d_in[15];
    const float* dct3_w   = (const float*)d_in[16];
    const float* dct3_b   = (const float*)d_in[17];
    float* outp = (float*)d_out;

    const int B = in_sizes[0] / 576;
    (void)n_in; (void)out_size; (void)ws_size;

    char* W = (char*)d_ws;
    size_t off = 0;
    auto take = [&](size_t bytes) -> char* {
        char* p = W + off;
        off = (off + bytes + 255) & ~(size_t)255;
        return p;
    };

    unsigned short* wp2  = (unsigned short*)take((size_t)9 * 64 * 32 * 2);
    unsigned short* wp3  = (unsigned short*)take((size_t)9 * 128 * 64 * 2);
    unsigned short* wd1  = (unsigned short*)take((size_t)9 * 64 * 128 * 2);
    unsigned short* wd2  = (unsigned short*)take((size_t)9 * 32 * 64 * 2);
    unsigned short* wfce = (unsigned short*)take((size_t)128 * 1152 * 2);
    unsigned short* wcb  = (unsigned short*)take((size_t)1024 * 128 * 2);
    float*          cnh  = (float*)take((size_t)1024 * 4);
    unsigned short* wfcd = (unsigned short*)take((size_t)1152 * 128 * 2);
    float*          bpd  = (float*)take((size_t)1152 * 4);
    unsigned short* zbf  = (unsigned short*)take((size_t)B * 128 * 2);
    int*            idx  = (int*)take((size_t)B * 4);
    unsigned short* zqbf = (unsigned short*)take((size_t)B * 128 * 2);
    unsigned short* act1 = (unsigned short*)take((size_t)B * 144 * 32 * 2); // also dct2 out
    unsigned short* act2 = (unsigned short*)take((size_t)B * 36 * 64 * 2);  // also dct1 out
    unsigned short* act3 = (unsigned short*)take((size_t)B * 9 * 128 * 2);  // also dec_fc out
    unsigned short* actD0 = act3;
    unsigned short* actD1 = act2;
    unsigned short* actD2 = act1;

    // ---- weight packing ----
    pack_conv_enc_k<<<72, 256, 0, stream>>>(conv2_w, wp2, 32, 64);
    pack_conv_enc_k<<<288, 256, 0, stream>>>(conv3_w, wp3, 64, 128);
    pack_conv_dec_k<<<288, 256, 0, stream>>>(dct1_w, wd1, 128, 64);
    pack_conv_dec_k<<<72, 256, 0, stream>>>(dct2_w, wd2, 64, 32);
    pack_enc_fc_k<<<576, 256, 0, stream>>>(enc_fc_w, wfce);
    pack_cb_k<<<512, 256, 0, stream>>>(codebook, wcb);
    cnorm_k<<<4, 256, 0, stream>>>(codebook, cnh);
    pack_dec_fc_k<<<576, 256, 0, stream>>>(dec_fc_w, wfcd);
    pack_dec_bias_k<<<5, 256, 0, stream>>>(dec_fc_b, bpd);

    // ---- encoder ----
    conv1_relu_pool<<<(B * 144 + 255) / 256, 256, 0, stream>>>(x, conv1_w, conv1_b, act1, B);
    conv3x3_relu_wmma<32, 64, 12, 1, true><<<B, 128, 0, stream>>>(act1, wp2, conv2_b, act2);
    conv3x3_relu_wmma<64, 128, 6, 4, true><<<B / 4, 128, 0, stream>>>(act2, wp3, conv3_b, act3);
    fc_wmma<<<B / 64, 128, 0, stream>>>(act3, wfce, enc_fc_b, zbf, 1152, 128);

    // ---- quantize ----
    vq_argmin_wmma<<<B / 64, 128, 0, stream>>>(zbf, wcb, cnh, idx);
    vq_gather<<<(B * 128 + 255) / 256, 256, 0, stream>>>(idx, codebook, zqbf, B * 128);

    // ---- decoder ----
    fc_wmma<<<B / 64, 128, 0, stream>>>(zqbf, wfcd, bpd, actD0, 128, 1152);
    conv3x3_relu_wmma<128, 64, 3, 4, false><<<B / 4, 128, 0, stream>>>(actD0, wd1, dct1_b, actD1);
    conv3x3_relu_wmma<64, 32, 6, 4, false><<<B / 4, 128, 0, stream>>>(actD1, wd2, dct2_b, actD2);
    dct3_relu_upsample<<<(B * 144 + 255) / 256, 256, 0, stream>>>(actD2, dct3_w, dct3_b, outp, B);
}
